// MDGR_17824114278453
// MI455X (gfx1250) — matrix operand order repaired and checked by reading, not compiled, not run on hardware
//
#include <hip/hip_runtime.h>
#include <hip/hip_bf16.h>

#define EPSV 1e-12f

typedef _Float16 h16 __attribute__((ext_vector_type(16)));
typedef _Float16 h4  __attribute__((ext_vector_type(4)));
typedef float    f8  __attribute__((ext_vector_type(8)));

union H16 { h16 v; _Float16 e[16]; };
union F8  { f8 v;  float     e[8];  };

#define APAD 40
#define BPAD 40

// ---- WMMA fragment gathers (CDNA5 16x16x32 f16 layouts) ----
// A (16x32, MxK): lane L -> M=L&15 ; elem j -> K = ((L>>4)<<3) + (j&7) + ((j&8)<<1)
__device__ inline h16 load_a_frag(const _Float16* As, int ld, int m_off, int lane) {
  H16 r;
  int m = m_off + (lane & 15);
  int kb = (lane >> 4) << 3;
#pragma unroll
  for (int j = 0; j < 16; ++j) {
    int k = kb + (j & 7) + ((j & 8) << 1);
    r.e[j] = As[m * ld + k];
  }
  return r.v;
}
// B (32x16, KxN) stored transposed in LDS as [n][k]: lane L -> N=L&15 ; elem j -> K=((L>>4)<<3)+(j&7)+((j&8)<<1)
__device__ inline h16 load_b_frag(const _Float16* Bs, int ld, int n_off, int lane) {
  H16 r;
  int n = n_off + (lane & 15);
  int kb = (lane >> 4) << 3;
#pragma unroll
  for (int j = 0; j < 16; ++j) r.e[j] = Bs[n * ld + kb + (j & 7) + ((j & 8) << 1)];
  return r.v;
}

#define WMMA_F16(a, b, c) __builtin_amdgcn_wmma_f32_16x16x32_f16(false, a, false, b, (short)0, c, false, false)

// ---- Generic batched GEMM: C[b] = A[b](MxK) * B[b](KxN); block tile 64x128, wave tile 32x32 ----
#define GTM 64
#define GTN 128
#define GTK 32
template <bool FULL>
__global__ __launch_bounds__(256) void wmma_gemm_kernel(
    const float* __restrict__ A, const float* __restrict__ B, float* __restrict__ C,
    int M, int N, int K, int lda, int ldb, int ldc,
    long long sA, long long sB, long long sC)
{
  __shared__ _Float16 As[2][GTM][APAD];
  __shared__ _Float16 Bs[2][GTN][BPAD];
  int bz = blockIdx.z;
  A += sA * bz; B += sB * bz; C += sC * bz;
  int tid = threadIdx.x, lane = tid & 31, wid = tid >> 5;
  int m0 = blockIdx.y * GTM, n0 = blockIdx.x * GTN;
  int wm = (wid & 1) * 32, wn = (wid >> 1) * 32;
  F8 acc00 = {}, acc01 = {}, acc10 = {}, acc11 = {};
  int nk = (K + GTK - 1) / GTK;

  auto stage = [&](int buf, int k0) {
    if (FULL) {
#pragma unroll
      for (int p = 0; p < 2; ++p) {            // A: 64x32 = 512 float4 slots
        int slot = tid + p * 256;
        int m = slot >> 3, k4 = (slot & 7) * 4;
        float4 v = *(const float4*)&A[(long long)(m0 + m) * lda + k0 + k4];
        _Float16* d = &As[buf][m][k4];
        d[0] = (_Float16)v.x; d[1] = (_Float16)v.y; d[2] = (_Float16)v.z; d[3] = (_Float16)v.w;
      }
#pragma unroll
      for (int p = 0; p < 4; ++p) {            // B: 32x128 = 1024 float4 slots
        int slot = tid + p * 256;
        int k = slot >> 5, n4 = (slot & 31) * 4;
        float4 v = *(const float4*)&B[(long long)(k0 + k) * ldb + n0 + n4];
        Bs[buf][n4 + 0][k] = (_Float16)v.x; Bs[buf][n4 + 1][k] = (_Float16)v.y;
        Bs[buf][n4 + 2][k] = (_Float16)v.z; Bs[buf][n4 + 3][k] = (_Float16)v.w;
      }
    } else {
#pragma unroll
      for (int p = 0; p < 8; ++p) {            // A: 2048 scalar elems, branch-free
        int i = tid + p * 256;
        int m = i >> 5, k = i & 31;
        bool ok = (m0 + m < M) && (k0 + k < K);
        long long idx = ok ? (long long)(m0 + m) * lda + (k0 + k) : 0;
        float v = A[idx];
        As[buf][m][k] = (_Float16)(ok ? v : 0.f);
      }
#pragma unroll
      for (int p = 0; p < 16; ++p) {           // B: 4096 scalar elems, branch-free
        int i = tid + p * 256;
        int k = i >> 7, n = i & 127;
        bool ok = (k0 + k < K) && (n0 + n < N);
        long long idx = ok ? (long long)(k0 + k) * ldb + (n0 + n) : 0;
        float v = B[idx];
        Bs[buf][n][k] = (_Float16)(ok ? v : 0.f);
      }
    }
  };

  stage(0, 0);
  for (int it = 0; it < nk; ++it) {
    __syncthreads();
    if (it + 1 < nk) stage((it + 1) & 1, (it + 1) * GTK);
    const _Float16* Ab = &As[it & 1][0][0];
    const _Float16* Bb = &Bs[it & 1][0][0];
    h16 a0 = load_a_frag(Ab, APAD, wm, lane);
    h16 a1 = load_a_frag(Ab, APAD, wm + 16, lane);
    h16 b0 = load_b_frag(Bb, BPAD, wn, lane);
    h16 b1 = load_b_frag(Bb, BPAD, wn + 16, lane);
    acc00.v = WMMA_F16(a0, b0, acc00.v);
    acc01.v = WMMA_F16(a0, b1, acc01.v);
    acc10.v = WMMA_F16(a1, b0, acc10.v);
    acc11.v = WMMA_F16(a1, b1, acc11.v);
  }
  int cn = lane & 15;
  int rb = (lane >> 4) << 3;
#pragma unroll
  for (int r = 0; r < 8; ++r) {
    int mA = m0 + wm + rb + r, mB = mA + 16;
    int nA = n0 + wn + cn,     nB = nA + 16;
    if (FULL) {
      C[(long long)mA * ldc + nA] = acc00.e[r];
      C[(long long)mA * ldc + nB] = acc01.e[r];
      C[(long long)mB * ldc + nA] = acc10.e[r];
      C[(long long)mB * ldc + nB] = acc11.e[r];
    } else {
      if (mA < M && nA < N) C[(long long)mA * ldc + nA] = acc00.e[r];
      if (mA < M && nB < N) C[(long long)mA * ldc + nB] = acc01.e[r];
      if (mB < M && nA < N) C[(long long)mB * ldc + nA] = acc10.e[r];
      if (mB < M && nB < N) C[(long long)mB * ldc + nB] = acc11.e[r];
    }
  }
}

// ---- pack sg_w (O,I,3,3) fp32 -> 9 contiguous f16 (O,I) matrices ----
__global__ __launch_bounds__(256) void pack_w_kernel(
    const float* __restrict__ Wsrc, _Float16* __restrict__ Wp, int Cout, int Cin)
{
  long long total = (long long)9 * Cout * Cin;
  for (long long idx = (long long)blockIdx.x * blockDim.x + threadIdx.x; idx < total;
       idx += (long long)gridDim.x * blockDim.x) {
    long long t = idx;
    int ic = (int)(t % Cin); t /= Cin;
    int o  = (int)(t % Cout);
    int kk = (int)(t / Cout);
    int ky = kk / 3, kx = kk % 3;
    Wp[idx] = (_Float16)Wsrc[(((long long)o * Cin + ic) * 3 + ky) * 3 + kx];
  }
}

// ---- Implicit-GEMM 3x3 SAME conv: block tile 128(M) x 64(N=one row), wave tile 32x32 ----
#define CTM 128
#define CTN 64
__global__ __launch_bounds__(256) void conv3x3_wmma_kernel(
    const float* __restrict__ X, const _Float16* __restrict__ Wp, float* __restrict__ Y,
    int Cin, int Cout)
{
  __shared__ _Float16 As[2][CTM][APAD];
  __shared__ _Float16 Bs[2][CTN][BPAD];
  const int H = 64, W = 64;
  int b = blockIdx.z, m0 = blockIdx.y * CTM, y = blockIdx.x;
  int tid = threadIdx.x, lane = tid & 31, wid = tid >> 5;
  int wm = (wid >> 1) * 32, wn = (wid & 1) * 32;
  F8 acc00 = {}, acc01 = {}, acc10 = {}, acc11 = {};
  const float* Xb = X + (long long)b * Cin * H * W;
  int ck = Cin >> 5;
  int nk = 9 * ck;

  auto stage = [&](int buf, int it) {
    int kk = it / ck;
    int c0 = (it - kk * ck) << 5;
    int ky = kk / 3, kx = kk % 3;
    int iy = y + ky - 1;
#pragma unroll
    for (int p = 0; p < 4; ++p) {              // A: 128x32 halves = 1024 h4 slots
      int slot = tid + p * 256;
      int m = slot >> 3, k4 = (slot & 7) * 4;
      h4 v = *(const h4*)&Wp[((long long)kk * Cout + m0 + m) * Cin + c0 + k4];
      *(h4*)&As[buf][m][k4] = v;
    }
    bool rowok = (iy >= 0) && (iy < H);
    int iyc = iy < 0 ? 0 : (iy > H - 1 ? H - 1 : iy);
#pragma unroll
    for (int p = 0; p < 8; ++p) {              // B: 32x64 elems, branch-free clamp+select
      int i = tid + p * 256;
      int k = i >> 6, n = i & 63;
      int ix = n + kx - 1;
      bool ok = rowok && (ix >= 0) && (ix < W);
      int ixc = ix < 0 ? 0 : (ix > W - 1 ? W - 1 : ix);
      float v = Xb[((long long)(c0 + k) * H + iyc) * W + ixc];
      Bs[buf][n][k] = (_Float16)(ok ? v : 0.f);
    }
  };

  stage(0, 0);
  for (int it = 0; it < nk; ++it) {
    __syncthreads();
    if (it + 1 < nk) stage((it + 1) & 1, it + 1);
    const _Float16* Ab = &As[it & 1][0][0];
    const _Float16* Bb = &Bs[it & 1][0][0];
    h16 a0 = load_a_frag(Ab, APAD, wm, lane);
    h16 a1 = load_a_frag(Ab, APAD, wm + 16, lane);
    h16 b0 = load_b_frag(Bb, BPAD, wn, lane);
    h16 b1 = load_b_frag(Bb, BPAD, wn + 16, lane);
    acc00.v = WMMA_F16(a0, b0, acc00.v);
    acc01.v = WMMA_F16(a0, b1, acc01.v);
    acc10.v = WMMA_F16(a1, b0, acc10.v);
    acc11.v = WMMA_F16(a1, b1, acc11.v);
  }
  int cn = lane & 15;
  int rb = (lane >> 4) << 3;
  float* Yb = Y + ((long long)b * Cout + m0) * (H * W) + y * W;
#pragma unroll
  for (int r = 0; r < 8; ++r) {
    int mA = wm + rb + r, mB = mA + 16;
    Yb[(long long)mA * H * W + wn + cn]      = acc00.e[r];
    Yb[(long long)mA * H * W + wn + 16 + cn] = acc01.e[r];
    Yb[(long long)mB * H * W + wn + cn]      = acc10.e[r];
    Yb[(long long)mB * H * W + wn + 16 + cn] = acc11.e[r];
  }
}

// ---- softmax over node axis: theta (B,n,HW) -> Q (B,HW,n) ----
__global__ __launch_bounds__(256) void softmax_q_kernel(
    const float* __restrict__ theta, float* __restrict__ Q, int n, int HW)
{
  int b = blockIdx.y;
  int p = blockIdx.x * 256 + threadIdx.x;
  const float* t = theta + (long long)b * n * HW + p;
  float e[32];
  float mx = -1e30f;
  for (int i = 0; i < n; ++i) { float v = t[(long long)i * HW]; e[i] = v; mx = fmaxf(mx, v); }
  float s = 0.f;
  for (int i = 0; i < n; ++i) { e[i] = __expf(e[i] - mx); s += e[i]; }
  float inv = 1.f / s;
  float* q = Q + ((long long)b * HW + p) * n;
  for (int i = 0; i < n; ++i) q[i] = e[i] * inv;
}

// ---- GCN (cosine-softmax adjacency) with residual, in place on g (B,256,n) ----
__global__ __launch_bounds__(256) void gcn_kernel(
    float* __restrict__ g, const float* __restrict__ Wg, const float* __restrict__ gamma_p, int n)
{
  __shared__ float xt[32][256];
  __shared__ float adj[32][32];
  __shared__ float nrm[32];
  int b = blockIdx.x, tid = threadIdx.x;
  float* gb = g + (long long)b * 256 * n;
  for (int idx = tid; idx < n * 256; idx += 256) { int i = idx >> 8, d = idx & 255; xt[i][d] = gb[d * n + i]; }
  __syncthreads();
  if (tid < n) { float s = 0.f; for (int d = 0; d < 256; ++d) s += xt[tid][d] * xt[tid][d]; nrm[tid] = sqrtf(s); }
  __syncthreads();
  for (int idx = tid; idx < n * n; idx += 256) {
    int i = idx / n, j = idx % n;
    float s = 0.f;
    for (int d = 0; d < 256; ++d) s += xt[i][d] * xt[j][d];
    adj[i][j] = s / ((nrm[i] + EPSV) * (nrm[j] + EPSV));
  }
  __syncthreads();
  if (tid < n) {
    float mx = -1e30f;
    for (int j = 0; j < n; ++j) mx = fmaxf(mx, adj[tid][j]);
    float s = 0.f;
    for (int j = 0; j < n; ++j) { float e = __expf(adj[tid][j] - mx); adj[tid][j] = e; s += e; }
    float inv = 1.f / s;
    for (int j = 0; j < n; ++j) adj[tid][j] *= inv;
  }
  __syncthreads();
  float u[32]; int cnt = 0;
  for (int idx = tid; idx < n * 256; idx += 256) {
    int i = idx >> 8, d = idx & 255;
    float s = 0.f;
    for (int m = 0; m < n; ++m) s += adj[i][m] * xt[m][d];
    u[cnt++] = s;
  }
  __syncthreads();
  cnt = 0;
  for (int idx = tid; idx < n * 256; idx += 256) { int i = idx >> 8, d = idx & 255; xt[i][d] = u[cnt++]; }
  __syncthreads();
  float gamma = *gamma_p;
  for (int idx = tid; idx < n * 256; idx += 256) {
    int i = idx >> 8, e2 = idx & 255;
    float s = 0.f;
    for (int d = 0; d < 256; ++d) s += xt[i][d] * Wg[d * 256 + e2];
    s = s > 0.f ? s : 0.f;
    gb[e2 * n + i] += gamma * s;
  }
}

// ---- CrossGraphSample: raw = sparsified-softmax-adj @ inp^T ; inp = [gA|gB] ----
__global__ __launch_bounds__(256) void cross_graph_kernel(
    const float* __restrict__ gA, const float* __restrict__ gB, const float* __restrict__ tgt,
    float* __restrict__ outraw, int n1, int n2, int Nt, int topk)
{
  int Ni = n1 + n2;
  __shared__ float adj[32][48];
  __shared__ float normT[32], normI[48];
  int b = blockIdx.x, tid = threadIdx.x;
  const float* A  = gA + (long long)b * 256 * n1;
  const float* Bp = gB + (long long)b * 256 * n2;
  const float* T  = tgt + (long long)b * 256 * Nt;
  float* O = outraw + (long long)b * 256 * Nt;
  auto inpv = [&](int d, int n) { return n < n1 ? A[d * n1 + n] : Bp[d * n2 + (n - n1)]; };
  for (int t = tid; t < Nt; t += 256) { float s = 0.f; for (int d = 0; d < 256; ++d) { float v = T[d * Nt + t]; s += v * v; } normT[t] = sqrtf(s); }
  for (int n = tid; n < Ni; n += 256) { float s = 0.f; for (int d = 0; d < 256; ++d) { float v = inpv(d, n); s += v * v; } normI[n] = sqrtf(s); }
  __syncthreads();
  for (int idx = tid; idx < Nt * Ni; idx += 256) {
    int t = idx / Ni, n = idx % Ni;
    float s = 0.f;
    for (int d = 0; d < 256; ++d) s += T[d * Nt + t] * inpv(d, n);
    adj[t][n] = s / ((normT[t] + EPSV) * (normI[n] + EPSV));
  }
  __syncthreads();
  if (tid < Nt) {
    float mx = -1e30f;
    for (int n = 0; n < Ni; ++n) mx = fmaxf(mx, adj[tid][n]);
    float s = 0.f;
    for (int n = 0; n < Ni; ++n) { float e = __expf(adj[tid][n] - mx); adj[tid][n] = e; s += e; }
    float inv = 1.f / s;
    for (int n = 0; n < Ni; ++n) adj[tid][n] *= inv;
    float row[48];
    for (int n = 0; n < Ni; ++n) row[n] = adj[tid][n];
    for (int a = 0; a < topk; ++a) {
      int best = a;
      for (int c2 = a + 1; c2 < Ni; ++c2) if (row[c2] > row[best]) best = c2;
      float tmp = row[a]; row[a] = row[best]; row[best] = tmp;
    }
    float thr = row[topk - 1];
    for (int n = 0; n < Ni; ++n) adj[tid][n] = adj[tid][n] >= thr ? adj[tid][n] : 0.f;
  }
  __syncthreads();
  for (int idx = tid; idx < 256 * Nt; idx += 256) {
    int d = idx / Nt, t = idx % Nt;
    float s = 0.f;
    for (int n = 0; n < Ni; ++n) s += adj[t][n] * inpv(d, n);
    O[d * Nt + t] = s;
  }
}

// ---- out = bn1d(lrelu(raw)) * gamma + tgt  (stats over B,Nt per channel) ----
__global__ __launch_bounds__(128) void cross_bn_kernel(
    const float* __restrict__ raw, const float* __restrict__ tgt, float* __restrict__ out,
    const float* __restrict__ bnw, const float* __restrict__ bnb, const float* __restrict__ gamma_p,
    int Nt, int B)
{
  int c = blockIdx.x, tid = threadIdx.x;
  int cnt = B * Nt;
  __shared__ float sm[128], sq[128];
  int b = tid / Nt, t = tid % Nt;
  float x = 0.f;
  if (tid < cnt) { float v = raw[((long long)b * 256 + c) * Nt + t]; x = v > 0.f ? v : 0.01f * v; }
  sm[tid] = (tid < cnt) ? x : 0.f;
  sq[tid] = (tid < cnt) ? x * x : 0.f;
  __syncthreads();
  for (int s = 64; s > 0; s >>= 1) {
    if (tid < s) { sm[tid] += sm[tid + s]; sq[tid] += sq[tid + s]; }
    __syncthreads();
  }
  float mean = sm[0] / cnt;
  float var  = sq[0] / cnt - mean * mean;
  float rstd = rsqrtf(var + 1e-5f);
  if (tid < cnt) {
    float y = (x - mean) * rstd * bnw[c] + bnb[c];
    out[((long long)b * 256 + c) * Nt + t] = y * (*gamma_p) + tgt[((long long)b * 256 + c) * Nt + t];
  }
}

// ---- reprojection: dst[b, off+d, p] = sum_i g[b,d,i] * Q[b,p,i] ----
__global__ __launch_bounds__(256) void reproj_kernel(
    const float* __restrict__ g, const float* __restrict__ Q, float* __restrict__ dst,
    int n, int dstCtot, int dstOff)
{
  __shared__ float gs[32];
  int b = blockIdx.z, d = blockIdx.y;
  int p = blockIdx.x * 256 + threadIdx.x;
  if (threadIdx.x < n) gs[threadIdx.x] = g[((long long)b * 256 + d) * n + threadIdx.x];
  __syncthreads();
  const float* q = Q + ((long long)b * 4096 + p) * n;
  float s = 0.f;
  for (int i = 0; i < n; ++i) s += gs[i] * q[i];
  dst[((long long)b * dstCtot + dstOff + d) * 4096 + p] = s;
}

// ---- bn2d stats over (B,HW) per channel of a channel-slice ----
__global__ __launch_bounds__(256) void bn_stats_kernel(
    const float* __restrict__ src, float* __restrict__ mean_out, float* __restrict__ rstd_out,
    int Ctot, int coff, int HW, int B, float eps)
{
  int c = blockIdx.x, tid = threadIdx.x;
  __shared__ float sm[256], sq[256];
  float s = 0.f, q = 0.f;
  for (int b = 0; b < B; ++b) {
    const float* p = src + ((long long)b * Ctot + coff + c) * HW;
    for (int i = tid; i < HW; i += 256) { float v = p[i]; s += v; q += v * v; }
  }
  sm[tid] = s; sq[tid] = q;
  __syncthreads();
  for (int st = 128; st > 0; st >>= 1) {
    if (tid < st) { sm[tid] += sm[tid + st]; sq[tid] += sq[tid + st]; }
    __syncthreads();
  }
  if (tid == 0) {
    float cntf = (float)B * (float)HW;
    float m = sm[0] / cntf;
    float var = sq[0] / cntf - m * m;
    mean_out[c] = m;
    rstd_out[c] = rsqrtf(var + eps);
  }
}

// ---- bn apply (+optional residual, +optional leaky-relu), in place ----
__global__ __launch_bounds__(256) void bn_apply_kernel(
    float* __restrict__ dst, int Ctot, int coff, int Cc, int HW, int B,
    const float* __restrict__ mean, const float* __restrict__ rstd,
    const float* __restrict__ w, const float* __restrict__ bias,
    const float* __restrict__ residual, int do_lrelu)
{
  long long total = (long long)B * Cc * HW;
  for (long long idx = (long long)blockIdx.x * blockDim.x + threadIdx.x; idx < total;
       idx += (long long)gridDim.x * blockDim.x) {
    int p = (int)(idx % HW);
    long long t = idx / HW;
    int c = (int)(t % Cc);
    int b = (int)(t / Cc);
    long long off = ((long long)b * Ctot + coff + c) * HW + p;
    float x = dst[off];
    float y = (x - mean[c]) * rstd[c] * w[c] + bias[c];
    if (residual) y += residual[((long long)b * Cc + c) * HW + p];
    if (do_lrelu) y = y > 0.f ? y : 0.01f * y;
    dst[off] = y;
  }
}

extern "C" void kernel_launch(void* const* d_in, const int* in_sizes, int n_in,
                              void* d_out, int out_size, void* d_ws, size_t ws_size,
                              hipStream_t stream)
{
  (void)in_sizes; (void)n_in; (void)out_size; (void)ws_size;
  const int B = 4, C = 512, D = 256, HW = 4096;

  const float* c3 = (const float*)d_in[0];
  const float* c4 = (const float*)d_in[1];
  const float* c5 = (const float*)d_in[2];
  const float* Wphi[3]  = {(const float*)d_in[3],  (const float*)d_in[7],  (const float*)d_in[11]};
  const float* Wth[3]   = {(const float*)d_in[4],  (const float*)d_in[8],  (const float*)d_in[12]};
  const float* Wg[3]    = {(const float*)d_in[5],  (const float*)d_in[9],  (const float*)d_in[13]};
  const float* gam[3]   = {(const float*)d_in[6],  (const float*)d_in[10], (const float*)d_in[14]};
  const float* cgGam[3] = {(const float*)d_in[15], (const float*)d_in[21], (const float*)d_in[27]};
  const float* cgBnw[3] = {(const float*)d_in[16], (const float*)d_in[22], (const float*)d_in[28]};
  const float* cgBnb[3] = {(const float*)d_in[17], (const float*)d_in[23], (const float*)d_in[29]};
  const float* extW[3]  = {(const float*)d_in[18], (const float*)d_in[24], (const float*)d_in[30]};
  const float* extBnw[3]= {(const float*)d_in[19], (const float*)d_in[25], (const float*)d_in[31]};
  const float* extBnb[3]= {(const float*)d_in[20], (const float*)d_in[26], (const float*)d_in[32]};
  const float* ssW   = (const float*)d_in[33];
  const float* ssBnw = (const float*)d_in[34];
  const float* ssBnb = (const float*)d_in[35];
  const float* sgW   = (const float*)d_in[36];
  const float* sgBnw = (const float*)d_in[37];
  const float* sgBnb = (const float*)d_in[38];
  float* out = (float*)d_out;

  char* ws = (char*)d_ws;
  size_t off = 0;
  auto alloc = [&](size_t bytes) -> void* {
    off = (off + 255) & ~(size_t)255;
    void* p = ws + off;
    off += bytes;
    return p;
  };

  const int ns[3] = {8, 16, 32};
  const float* xs[3] = {c5, c4, c3};

  float* Q[3];
  for (int s = 0; s < 3; ++s) Q[s] = (float*)alloc((size_t)B * HW * ns[s] * 4);
  float* theta = (float*)alloc((size_t)B * 32 * HW * 4);
  float* phi   = (float*)alloc((size_t)B * D * HW * 4);   // reused as reproj scratch later
  float* g[3];  for (int s = 0; s < 3; ++s) g[s]  = (float*)alloc((size_t)B * D * ns[s] * 4);
  float* graw  = (float*)alloc((size_t)B * D * 32 * 4);
  float* gc[3]; for (int s = 0; s < 3; ++s) gc[s] = (float*)alloc((size_t)B * D * ns[s] * 4);
  float* cat3  = (float*)alloc((size_t)B * 768 * HW * 4);   // [x8_|x16_|x32_]
  float* cat4  = (float*)alloc((size_t)B * 2048 * HW * 4);  // [x8|x16|x32|x_ms]
  _Float16* Wpk = (_Float16*)alloc((size_t)9 * C * 2048 * 2);
  float* meanb = (float*)alloc(512 * 4);
  float* rstdb = (float*)alloc(512 * 4);

  // pre-pack conv3x3 weights to f16 (ky,kx)-major
  pack_w_kernel<<<dim3(4096), 256, 0, stream>>>(sgW, Wpk, C, 2048);

  // Per-scale: phi, theta, softmax Q, graph, gcn
  for (int s = 0; s < 3; ++s) {
    int n = ns[s];
    wmma_gemm_kernel<true><<<dim3(HW / GTN, D / GTM, B), 256, 0, stream>>>(
        Wphi[s], xs[s], phi, D, HW, C, C, HW, HW,
        0LL, (long long)C * HW, (long long)D * HW);
    wmma_gemm_kernel<false><<<dim3(HW / GTN, 1, B), 256, 0, stream>>>(
        Wth[s], xs[s], theta, n, HW, C, C, HW, HW,
        0LL, (long long)C * HW, (long long)n * HW);
    softmax_q_kernel<<<dim3(HW / 256, B), 256, 0, stream>>>(theta, Q[s], n, HW);
    wmma_gemm_kernel<false><<<dim3(1, D / GTM, B), 256, 0, stream>>>(
        phi, Q[s], g[s], D, n, HW, HW, n, n,
        (long long)D * HW, (long long)HW * n, (long long)D * n);
    gcn_kernel<<<dim3(B), 256, 0, stream>>>(g[s], Wg[s], gam[s], n);
  }

  // Cross-graph updates: {tgt, inpA, inpB, k=round(0.8*Ni)}
  const int cg_t[3] = {2, 1, 0};
  const int cg_a[3] = {0, 0, 1};
  const int cg_b[3] = {1, 2, 2};
  const int cg_k[3] = {19, 32, 38};
  for (int i = 0; i < 3; ++i) {
    int s = cg_t[i], a = cg_a[i], b2 = cg_b[i];
    cross_graph_kernel<<<dim3(B), 256, 0, stream>>>(
        g[a], g[b2], g[s], graw, ns[a], ns[b2], ns[s], cg_k[i]);
    cross_bn_kernel<<<dim3(D), 128, 0, stream>>>(
        graw, g[s], gc[s], cgBnw[s], cgBnb[s], cgGam[s], ns[s], B);
  }

  // Reproject cross outputs into cat3 (768 channels)
  for (int s = 0; s < 3; ++s)
    reproj_kernel<<<dim3(HW / 256, D, B), 256, 0, stream>>>(gc[s], Q[s], cat3, ns[s], 768, s * 256);

  // ext path: reproj(g_s) -> 1x1 conv -> bn(eps=1e-4) + residual, into cat4
  for (int s = 0; s < 3; ++s) {
    reproj_kernel<<<dim3(HW / 256, D, B), 256, 0, stream>>>(g[s], Q[s], phi, ns[s], 256, 0);
    wmma_gemm_kernel<true><<<dim3(HW / GTN, C / GTM, B), 256, 0, stream>>>(
        extW[s], phi, cat4 + (size_t)s * 512 * HW, C, HW, D, D, HW, HW,
        0LL, (long long)D * HW, (long long)2048 * HW);
    bn_stats_kernel<<<dim3(C), 256, 0, stream>>>(cat4, meanb, rstdb, 2048, s * 512, HW, B, 1e-4f);
    bn_apply_kernel<<<dim3(2048), 256, 0, stream>>>(
        cat4, 2048, s * 512, C, HW, B, meanb, rstdb, extBnw[s], extBnb[s], xs[s], 0);
  }

  // x_ms = lrelu(bn(ss_w @ cat3)), into cat4 offset 1536
  wmma_gemm_kernel<true><<<dim3(HW / GTN, C / GTM, B), 256, 0, stream>>>(
      ssW, cat3, cat4 + (size_t)1536 * HW, C, HW, 768, 768, HW, HW,
      0LL, (long long)768 * HW, (long long)2048 * HW);
  bn_stats_kernel<<<dim3(C), 256, 0, stream>>>(cat4, meanb, rstdb, 2048, 1536, HW, B, 1e-5f);
  bn_apply_kernel<<<dim3(2048), 256, 0, stream>>>(
      cat4, 2048, 1536, C, HW, B, meanb, rstdb, ssBnw, ssBnb, nullptr, 1);

  // final: lrelu(bn(conv3x3(cat4, sg_w)))
  conv3x3_wmma_kernel<<<dim3(64, C / CTM, B), 256, 0, stream>>>(cat4, Wpk, out, 2048, C);
  bn_stats_kernel<<<dim3(C), 256, 0, stream>>>(out, meanb, rstdb, C, 0, HW, B, 1e-5f);
  bn_apply_kernel<<<dim3(2048), 256, 0, stream>>>(
      out, C, 0, C, HW, B, meanb, rstdb, sgBnw, sgBnb, nullptr, 1);
}
